// minGRU_79147657331406
// MI455X (gfx1250) — compile-verified, matrix operand order checked by reading, not run on hardware
//
#include <hip/hip_runtime.h>
#include <hip/hip_bf16.h>

// ---------------------------------------------------------------------------
// minGRU forward for MI455X (gfx1250, wave32, WMMA)
//   proj = x @ w_f + b_f                 (bf16 WMMA, f32 accum)
//   log-space chunked parallel scan      (f32)
//   h    = h_t @ w_down                  (bf16 WMMA, f32 accum)
// out1 = h [B,T,D] f32 ; out2 = log_h_t[:, 2:-1:3] [B,1365,E] f32
// ---------------------------------------------------------------------------

#define Bq   4
#define Tq   4096
#define Dq   1024
#define Eq   2048
#define E2q  (2 * Eq)      // 4096
#define Mq   (Bq * Tq)     // 16384 rows for both GEMMs
#define Cq   64            // chunks along T
#define Lq   (Tq / Cq)     // 64 steps per chunk
#define TSq  1365          // len(range(2, 4095, 3))

typedef __attribute__((ext_vector_type(16))) __bf16    v16bf;
typedef __attribute__((ext_vector_type(8)))  float     v8f;
typedef __attribute__((ext_vector_type(4)))  unsigned  v4u;

union BfPack { v4u u[2]; v16bf b; };

// ---------------- helpers ----------------

__device__ __forceinline__ unsigned short f2bf(float f) {
    unsigned u = __float_as_uint(f);
    u += 0x7fffu + ((u >> 16) & 1u);     // round-to-nearest-even
    return (unsigned short)(u >> 16);
}

__device__ __forceinline__ float softplusf(float x) {
    return x > 0.f ? x + log1pf(__expf(-x)) : log1pf(__expf(x));
}

__device__ __forceinline__ float logaddexpf_(float a, float b) {
    float mx = fmaxf(a, b), mn = fminf(a, b);
    if (mn == -INFINITY) return mx;       // covers (-inf,-inf) -> -inf
    return mx + log1pf(__expf(mn - mx));
}

__device__ __forceinline__ float log_gf(float x) {
    return x >= 0.f ? __logf(x + 0.5f) : -softplusf(-x);
}

// one recurrence step in log space: a = log_coeff, v = log_z + log_g(h_x)
__device__ __forceinline__ void step_av(float kv, float hx, float& a, float& v) {
    a = -softplusf(kv);                   // log(1 - sigmoid(k))
    float log_z = -softplusf(-kv);        // log sigmoid(k)
    v = log_z + log_gf(hx);
}

// load 16 bf16 (one WMMA A/B operand slice for this lane) as two b128s
__device__ __forceinline__ v16bf load16(const unsigned short* p) {
    BfPack pk;
    pk.u[0] = *(const v4u*)(p);
    pk.u[1] = *(const v4u*)(p + 16);
    return pk.b;
}

// ---------------- conversion kernels ----------------

__global__ void cvt_f32_to_bf16(const float* __restrict__ in,
                                unsigned short* __restrict__ out, int n) {
    int i = blockIdx.x * blockDim.x + threadIdx.x;
    if (i < n) out[i] = f2bf(in[i]);
}

// in: f32 [R, C] row-major  ->  out: bf16 [C, R] row-major (coalesced writes)
__global__ void cvt_transpose_bf16(const float* __restrict__ in,
                                   unsigned short* __restrict__ out,
                                   int R, int C) {
    int o = blockIdx.x * blockDim.x + threadIdx.x;
    if (o >= R * C) return;
    int c = o / R;
    int r = o - c * R;
    out[o] = f2bf(in[(size_t)r * C + c]);
}

// ---------------- WMMA GEMM: C[M,N] = A[M,K] * B^T[N,K] (+ bias[N]) --------
// One wave computes a 32x32 output block with 4 x v_wmma_f32_16x16x32_bf16.
// wave32 operand layout: lane l -> row/col (l&15); K-halves selected by l>>4.

__global__ void wmma_gemm_bf16(const unsigned short* __restrict__ A,
                               const unsigned short* __restrict__ Bt,
                               const float* __restrict__ bias,
                               float* __restrict__ C,
                               int M, int N, int K, int tilesN) {
    int wave = blockIdx.x * (blockDim.x >> 5) + (threadIdx.x >> 5);
    int lane = threadIdx.x & 31;
    int tm = wave / tilesN;
    int tn = wave - tm * tilesN;
    int mBase = tm * 32, nBase = tn * 32;
    if (mBase >= M) return;                      // wave-uniform: EXEC stays full

    int rsel = lane & 15;
    int hi   = (lane >> 4) * 8;                  // K sub-half offset

    const unsigned short* arow0 = A  + (size_t)(mBase + rsel)       * K + hi;
    const unsigned short* arow1 = A  + (size_t)(mBase + 16 + rsel)  * K + hi;
    const unsigned short* brow0 = Bt + (size_t)(nBase + rsel)       * K + hi;
    const unsigned short* brow1 = Bt + (size_t)(nBase + 16 + rsel)  * K + hi;

    v8f acc00 = {}, acc01 = {}, acc10 = {}, acc11 = {};

    for (int k = 0; k < K; k += 32) {
        v16bf a0 = load16(arow0 + k);
        v16bf a1 = load16(arow1 + k);
        v16bf b0 = load16(brow0 + k);
        v16bf b1 = load16(brow1 + k);
        __builtin_prefetch(arow0 + k + 128, 0, 1);   // global_prefetch_b8
        __builtin_prefetch(brow0 + k + 128, 0, 1);
        acc00 = __builtin_amdgcn_wmma_f32_16x16x32_bf16(false, a0, false, b0,
                                                        (short)0, acc00, false, false);
        acc01 = __builtin_amdgcn_wmma_f32_16x16x32_bf16(false, a0, false, b1,
                                                        (short)0, acc01, false, false);
        acc10 = __builtin_amdgcn_wmma_f32_16x16x32_bf16(false, a1, false, b0,
                                                        (short)0, acc10, false, false);
        acc11 = __builtin_amdgcn_wmma_f32_16x16x32_bf16(false, a1, false, b1,
                                                        (short)0, acc11, false, false);
    }

    float bv0 = bias ? bias[nBase + rsel]      : 0.f;
    float bv1 = bias ? bias[nBase + 16 + rsel] : 0.f;

    // D layout: lane l -> col (l&15); VGPR r -> row r + 8*(l>>4)
    int mrow = mBase + 8 * (lane >> 4);
    for (int r = 0; r < 8; ++r) {
        C[(size_t)(mrow + r)      * N + nBase + rsel]      = acc00[r] + bv0;
        C[(size_t)(mrow + r)      * N + nBase + 16 + rsel] = acc01[r] + bv1;
        C[(size_t)(mrow + 16 + r) * N + nBase + rsel]      = acc10[r] + bv0;
        C[(size_t)(mrow + 16 + r) * N + nBase + 16 + rsel] = acc11[r] + bv1;
    }
}

// ---------------- chunked log-space scan -----------------------------------
// thread id -> (b, c, e); proj[(b*T+t)*2E + e] = gate k, ... + e + E = h_x.

__global__ void scan1_chunk_agg(const float* __restrict__ proj,
                                float* __restrict__ aggA,
                                float* __restrict__ aggV) {
    int tid = blockIdx.x * blockDim.x + threadIdx.x;
    if (tid >= Bq * Cq * Eq) return;
    int e = tid & (Eq - 1);
    int rest = tid >> 11;          // / 2048
    int c = rest & (Cq - 1);
    int b = rest >> 6;

    float A = 0.f, V = -INFINITY;
    const float* pk = proj + ((size_t)(b * Tq + c * Lq)) * E2q + e;
    for (int i = 0; i < Lq; ++i, pk += E2q) {
        float a, v;
        step_av(pk[0], pk[Eq], a, v);
        A += a;
        V = logaddexpf_(V + a, v);
    }
    size_t idx = ((size_t)b * Cq + c) * Eq + e;   // [B,C,E] (coalesced both passes)
    aggA[idx] = A;
    aggV[idx] = V;
}

__global__ void scan2_cross_chunk(const float* __restrict__ log_h0,
                                  const float* __restrict__ aggA,
                                  const float* __restrict__ aggV,
                                  float* __restrict__ prefix) {
    int tid = blockIdx.x * blockDim.x + threadIdx.x;
    if (tid >= Bq * Eq) return;
    int e = tid & (Eq - 1);
    int b = tid >> 11;
    float r = log_h0[tid];
    for (int c = 0; c < Cq; ++c) {
        size_t idx = ((size_t)b * Cq + c) * Eq + e;
        prefix[idx] = r;
        r = logaddexpf_(r + aggA[idx], aggV[idx]);
    }
}

__global__ void scan3_replay(const float* __restrict__ proj,
                             const float* __restrict__ prefix,
                             unsigned short* __restrict__ h_bf,
                             float* __restrict__ out2) {
    int tid = blockIdx.x * blockDim.x + threadIdx.x;
    if (tid >= Bq * Cq * Eq) return;
    int e = tid & (Eq - 1);
    int rest = tid >> 11;
    int c = rest & (Cq - 1);
    int b = rest >> 6;

    float r = prefix[((size_t)b * Cq + c) * Eq + e];
    int t = c * Lq;
    const float* pk = proj + ((size_t)(b * Tq + t)) * E2q + e;
    for (int i = 0; i < Lq; ++i, ++t, pk += E2q) {
        float a, v;
        step_av(pk[0], pk[Eq], a, v);
        r = logaddexpf_(r + a, v);
        h_bf[((size_t)(b * Tq + t)) * Eq + e] = f2bf(__expf(r));
        if (t >= 2 && t < Tq - 1 && ((t - 2) % 3) == 0) {
            out2[((size_t)b * TSq + (t - 2) / 3) * Eq + e] = r;
        }
    }
}

// ---------------- launch -----------------------------------------------------

extern "C" void kernel_launch(void* const* d_in, const int* in_sizes, int n_in,
                              void* d_out, int out_size, void* d_ws, size_t ws_size,
                              hipStream_t stream) {
    const float* x       = (const float*)d_in[0];   // [B,T,D]
    const float* log_h0  = (const float*)d_in[1];   // [B,1,E]
    const float* w_f     = (const float*)d_in[2];   // [D,2E]
    const float* b_f     = (const float*)d_in[3];   // [2E]
    const float* w_down  = (const float*)d_in[4];   // [E,D]

    float* out1 = (float*)d_out;                              // [B,T,D]
    float* out2 = out1 + (size_t)Mq * Dq;                     // [B,1365,E]

    // workspace carve-up (all 256B-aligned by construction)
    char* ws = (char*)d_ws;
    unsigned short* x_bf  = (unsigned short*)(ws);                 //  32 MiB
    unsigned short* wfT   = (unsigned short*)(ws + 33554432);      //   8 MiB  [2E,D]
    unsigned short* wdT   = (unsigned short*)(ws + 41943040);      //   4 MiB  [D,E]
    float*          proj  = (float*)        (ws + 46137344);       // 256 MiB  [M,2E]
    float*          aggA  = (float*)        (ws + 314572800);      //   2 MiB  [B,C,E]
    float*          aggV  = (float*)        (ws + 316669952);      //   2 MiB
    float*          pref  = (float*)        (ws + 318767104);      //   2 MiB
    unsigned short* h_bf  = (unsigned short*)(ws + 320864256);     //  64 MiB  [M,E]

    const int TPB = 256;

    // 1) x -> bf16
    cvt_f32_to_bf16<<<(Mq * Dq + TPB - 1) / TPB, TPB, 0, stream>>>(x, x_bf, Mq * Dq);
    // 2) w_f [D,2E] -> wfT bf16 [2E,D]
    cvt_transpose_bf16<<<(Dq * E2q + TPB - 1) / TPB, TPB, 0, stream>>>(w_f, wfT, Dq, E2q);
    // 3) w_down [E,D] -> wdT bf16 [D,E]
    cvt_transpose_bf16<<<(Eq * Dq + TPB - 1) / TPB, TPB, 0, stream>>>(w_down, wdT, Eq, Dq);

    // 4) proj = x @ w_f + b_f  :  M=16384, N=4096, K=1024 -> 512x128 tiles of 32
    {
        int tilesM = Mq / 32, tilesN = E2q / 32;
        int waves = tilesM * tilesN;           // 65536
        wmma_gemm_bf16<<<waves / 8, TPB, 0, stream>>>(x_bf, wfT, b_f, proj,
                                                      Mq, E2q, Dq, tilesN);
    }

    // 5-7) chunked log-space scan
    scan1_chunk_agg<<<(Bq * Cq * Eq) / TPB, TPB, 0, stream>>>(proj, aggA, aggV);
    scan2_cross_chunk<<<(Bq * Eq) / TPB, TPB, 0, stream>>>(log_h0, aggA, aggV, pref);
    scan3_replay<<<(Bq * Cq * Eq) / TPB, TPB, 0, stream>>>(proj, pref, h_bf, out2);

    // 8) out1 = h_t @ w_down : M=16384, N=1024, K=2048 -> 512x32 tiles of 32
    {
        int tilesM = Mq / 32, tilesN = Dq / 32;
        int waves = tilesM * tilesN;           // 16384
        wmma_gemm_bf16<<<waves / 8, TPB, 0, stream>>>(h_bf, wdT, nullptr, out1,
                                                      Mq, Dq, Eq, tilesN);
    }
}
